// Circulant_89404039233568
// MI455X (gfx1250) — matrix operand order repaired and checked by reading, not compiled, not run on hardware
//
#include <hip/hip_runtime.h>

#define NN    4096
#define BATCH 1024
#define BK    32
#define BKP   36           // padded row stride (floats) to stagger LDS banks
#define BM    128          // 8 waves * 16 rows
#define BN    128          // 8 n-tiles of 16 per wave
#define NT    8            // n-tiles per wave
#define NK    (NN / BK)    // 128 K-steps

typedef __attribute__((ext_vector_type(16))) __bf16 v16bf;
typedef __attribute__((ext_vector_type(8)))  float  v8f;
typedef int v4i __attribute__((vector_size(16)));
typedef __attribute__((address_space(1))) v4i* gptr_v4i;
typedef __attribute__((address_space(3))) v4i* lptr_v4i;

#if defined(__has_builtin)
#if __has_builtin(__builtin_amdgcn_global_load_async_to_lds_b128)
#define HAVE_ASYNC_LDS 1
#endif
#endif

__device__ __forceinline__ unsigned short f2bf(float f) {
  union { float f; unsigned u; } x; x.f = f;
  unsigned r = x.u + 0x7fffu + ((x.u >> 16) & 1u);   // round-to-nearest-even
  return (unsigned short)(r >> 16);
}

// pack two floats -> two bf16 in one dword: round-half-up then v_perm_b32
// pool = {S0 bytes 4-7, S1 bytes 0-3}; sel 0x07060302 -> {hi[31:16], lo[31:16]}
__device__ __forceinline__ unsigned pk2bf(float lo, float hi) {
  const unsigned ul = __builtin_bit_cast(unsigned, lo) + 0x8000u;
  const unsigned uh = __builtin_bit_cast(unsigned, hi) + 0x8000u;
  return __builtin_amdgcn_perm(uh, ul, 0x07060302u);
}

union Frag16 {
  unsigned short s[16];
  unsigned u[8];
  v16bf v;
};

__global__ __launch_bounds__(256) void circulant_wmma(
    const float* __restrict__ x, const float* __restrict__ w,
    const float* __restrict__ bias, float* __restrict__ out) {
  __shared__ unsigned short w_lds[2 * NN];        // bf16 weight ring (mod-free indexing)
  __shared__ float x_lds[2][BM][BKP];             // double-buffered fp32 X stage

  const int tid  = threadIdx.x;
  const int lane = tid & 31;
  const int wid  = tid >> 5;
  const int hi   = lane >> 4;     // half-wave select
  const int l16  = lane & 15;
  const int gm0  = blockIdx.y * BM;
  const int gc0  = blockIdx.x * BN;

  // Stage one 128x32 fp32 chunk of X into x_lds[buf]; 4 x b128 per thread.
  auto stage = [&](int buf, int kt) {
    const int k0 = kt * BK;
#if HAVE_ASYNC_LDS
#pragma unroll
    for (int i = 0; i < 4; ++i) {
      const int f = tid + i * 256;            // flat 16B-quad index 0..1023
      const int row = f >> 3;
      const int q   = f & 7;
      const float* g = x + (size_t)(gm0 + row) * NN + k0 + q * 4;
      float* l = &x_lds[buf][row][q * 4];
      __builtin_amdgcn_global_load_async_to_lds_b128(
          (gptr_v4i)g, (lptr_v4i)l, 0, 0);
    }
#else
#pragma unroll
    for (int i = 0; i < 4; ++i) {
      const int f = tid + i * 256;
      const int row = f >> 3;
      const int q   = f & 7;
      const float4 v = *(const float4*)(x + (size_t)(gm0 + row) * NN + k0 + q * 4);
      *(float4*)(&x_lds[buf][row][q * 4]) = v;
    }
#endif
  };

  // Stage bf16 weight ring once per block (RNE; one-time cost).
  for (int j = tid; j < NN; j += 256) {
    const unsigned short b = f2bf(w[j]);
    w_lds[j]      = b;
    w_lds[j + NN] = b;
  }

  stage(0, 0);

  v8f acc[NT] = {};

  for (int kt = 0; kt < NK; ++kt) {
    const int cur = kt & 1;
#if HAVE_ASYNC_LDS
#if __has_builtin(__builtin_amdgcn_s_wait_asynccnt)
    __builtin_amdgcn_s_wait_asynccnt(0);
#else
    asm volatile("s_wait_asynccnt 0" ::: "memory");
#endif
#endif
    __syncthreads();
    if (kt + 1 < NK) stage(cur ^ 1, kt + 1);

    // ---- A fragment (16x32 bf16, contiguous k per lane half):
    // element e -> K = 16*hi + e ; row = wid*16 + l16
    Frag16 a;
    const float* xr = &x_lds[cur][wid * 16 + l16][hi * 16];
#pragma unroll
    for (int p = 0; p < 4; ++p) a.u[p] = pk2bf(xr[2 * p], xr[2 * p + 1]);
#pragma unroll
    for (int p = 0; p < 4; ++p) a.u[4 + p] = pk2bf(xr[8 + 2 * p], xr[8 + 2 * p + 1]);

    // ---- B fragments: B[r,c] = w[(r - c) mod N]; element e -> K = 16*hi + e
    const int kbase = kt * BK + 16 * hi + NN;   // +NN: mod-free ring index
#pragma unroll
    for (int t = 0; t < NT; ++t) {
      const int col = gc0 + t * 16 + l16;
      const unsigned short* wp = &w_lds[kbase - col];
      Frag16 b;
#pragma unroll
      for (int e = 0; e < 16; ++e) b.s[e] = wp[e];
      acc[t] = __builtin_amdgcn_wmma_f32_16x16x32_bf16(
          /*neg_a=*/false, a.v, /*neg_b=*/false, b.v,
          /*c_mod=*/(short)0, acc[t], /*reuse_a=*/false, /*reuse_b=*/false);
    }
  }

  // ---- Epilogue: C/D layout — VGPR v -> row (v + 8*hi), lane%16 -> col
#pragma unroll
  for (int t = 0; t < NT; ++t) {
    const int col = gc0 + t * 16 + l16;
    const float bv = bias[col];
#pragma unroll
    for (int v = 0; v < 8; ++v) {
      const int row = gm0 + wid * 16 + hi * 8 + v;
      __builtin_nontemporal_store(acc[t][v] + bv, &out[(size_t)row * NN + col]);
    }
  }
}

extern "C" void kernel_launch(void* const* d_in, const int* in_sizes, int n_in,
                              void* d_out, int out_size, void* d_ws, size_t ws_size,
                              hipStream_t stream) {
  const float* x    = (const float*)d_in[0];
  const float* w    = (const float*)d_in[1];
  const float* bias = (const float*)d_in[2];
  float* out        = (float*)d_out;
  dim3 grid(NN / BN, BATCH / BM);   // (32, 8) blocks
  circulant_wmma<<<grid, 256, 0, stream>>>(x, w, bias, out);
}
